// GAT_62414464746094
// MI455X (gfx1250) — matrix operand order repaired and checked by reading, not compiled, not run on hardware
//
#include <hip/hip_runtime.h>
#include <hip/hip_bf16.h>
#include <math.h>

typedef __attribute__((ext_vector_type(16))) __bf16 v16bf;
typedef __attribute__((ext_vector_type(8)))  __bf16 v8bf;
typedef __attribute__((ext_vector_type(8)))  float  v8f;

#define GAT_EPS 1e-16f
#define CDIV(a, b) (((a) + (b) - 1) / (b))

__device__ __forceinline__ __bf16 f32_to_bf16(float f) {
  unsigned u = __float_as_uint(f);
  u += 0x7fffu + ((u >> 16) & 1u);           // round-to-nearest-even
  unsigned short hi = (unsigned short)(u >> 16);
  __bf16 b;
  __builtin_memcpy(&b, &hi, sizeof(b));
  return b;
}

// order-preserving f32 <-> i32 encoding for integer atomicMax
__device__ __forceinline__ int enc_f32(float f) {
  int i = __float_as_int(f);
  return i >= 0 ? i : (i ^ 0x7fffffff);
}
__device__ __forceinline__ float dec_f32(int i) {
  return __int_as_float(i >= 0 ? i : (i ^ 0x7fffffff));
}

// ---------------- utility kernels ----------------

__global__ void gat_fill_u32(unsigned* __restrict__ p, int n, unsigned v) {
  int i = blockIdx.x * blockDim.x + threadIdx.x;
  if (i < n) p[i] = v;
}

__global__ void gat_cvt_bf16(const float* __restrict__ s, __bf16* __restrict__ d, int n) {
  int i = blockIdx.x * blockDim.x + threadIdx.x;
  if (i < n) d[i] = f32_to_bf16(s[i]);
}

// Wt[n*K + k] = (bf16) W[k*N + n]  (pre-transpose weights for contiguous B-fragment loads)
__global__ void gat_cvt_transpose(const float* __restrict__ W, __bf16* __restrict__ Wt,
                                  int K, int N) {
  int i = blockIdx.x * blockDim.x + threadIdx.x;
  if (i >= K * N) return;
  int k = i / N, n = i % N;
  Wt[(size_t)n * K + k] = f32_to_bf16(W[i]);
}

// ---------------- WMMA GEMM: C[M,N] = A[M,K] * B[K,N], Bt stored [N,K] ----------------
// One wave32 per 16-row strip, NT column tiles of 16 (NT*16 == N).

template <int NT>
__global__ void gat_wmma_gemm(const __bf16* __restrict__ A,
                              const __bf16* __restrict__ Bt,
                              float* __restrict__ C,
                              int M, int N, int K) {
  const int lane = threadIdx.x & 31;
  const int wave = threadIdx.x >> 5;
  const int strip = (blockIdx.x * (blockDim.x >> 5) + wave) * 16;
  if (strip >= M) return;

  const int mlo  = lane & 15;   // row within strip (both half-waves cover M=0..15)
  const int half = lane >> 4;   // K-interleave select

  v8f acc[NT] = {};

  const __bf16* arow = A + (size_t)(strip + mlo) * K;

  for (int kb = 0; kb < K; kb += 32) {
    // A fragment, contiguous k order (same as B):
    //   elems 0..15 -> K = kb + half*16 + 0..15
    union { v16bf v; v8bf h[2]; } a;
    a.v = *(const v16bf*)(arow + kb + half * 16);

#pragma unroll
    for (int t = 0; t < NT; ++t) {
      // B fragment: column n = t*16 + mlo, K slice = kb + half*16 + 0..15 (contiguous in Bt)
      const v16bf b = *(const v16bf*)(Bt + (size_t)(t * 16 + mlo) * K + kb + half * 16);
      acc[t] = __builtin_amdgcn_wmma_f32_16x16x32_bf16(
          false, a.v, false, b, (short)0, acc[t], false, false);
    }
  }

#pragma unroll
  for (int t = 0; t < NT; ++t) {
#pragma unroll
    for (int r = 0; r < 8; ++r) {
      // C/D layout: lane half 0 -> rows 0..7, half 1 -> rows 8..15
      C[(size_t)(strip + half * 8 + r) * N + t * 16 + mlo] = acc[t][r];
    }
  }
}

// ---------------- per-node attention projections ----------------
// hl[n,h] = sum_f al[h,f]*hfeat[n,h,f];  hr likewise. one thread per (node,head)

__global__ void gat_proj_lr(const float* __restrict__ h, const float* __restrict__ al,
                            const float* __restrict__ ar, float* __restrict__ hl,
                            float* __restrict__ hr, int total, int H, int F) {
  int i = blockIdx.x * blockDim.x + threadIdx.x;  // node*H + head
  if (i >= total) return;
  int head = i % H;
  const float* hp  = h + (size_t)i * F;
  const float* alp = al + head * F;
  const float* arp = ar + head * F;
  float sl = 0.f, sr = 0.f;
  for (int f = 0; f < F; ++f) {
    float v = hp[f];
    sl += alp[f] * v;
    sr += arp[f] * v;
  }
  hl[i] = sl;
  hr[i] = sr;
}

// ---------------- edge passes ----------------

__global__ void gat_edge_max(const int* __restrict__ row, const int* __restrict__ col,
                             const float* __restrict__ hl, const float* __restrict__ hr,
                             int* __restrict__ m_enc, int total, int H, float alpha) {
  int i = blockIdx.x * blockDim.x + threadIdx.x;  // edge*H + head
  if (i >= total) return;
  int e = i / H, h = i % H;
  int r = row[e], c = col[e];
  float v = hl[r * H + h] + hr[c * H + h];
  v = v > 0.f ? v : alpha * v;                    // leaky relu
  atomicMax(&m_enc[r * H + h], enc_f32(v));
}

__global__ void gat_edge_exp(const int* __restrict__ row, const int* __restrict__ col,
                             const float* __restrict__ hl, const float* __restrict__ hr,
                             const int* __restrict__ m_enc, float* __restrict__ s,
                             float* __restrict__ ew, int total, int H, float alpha) {
  int i = blockIdx.x * blockDim.x + threadIdx.x;
  if (i >= total) return;
  int e = i / H, h = i % H;
  int r = row[e], c = col[e];
  float v = hl[r * H + h] + hr[c * H + h];
  v = v > 0.f ? v : alpha * v;
  float m = dec_f32(m_enc[r * H + h]);
  float w = __expf(v - m);
  ew[i] = w;
  atomicAdd(&s[r * H + h], w);
}

// out[row[e], j] += (ew[e,head]/(s[row[e],head]+eps)) * hfeat[col[e], j]; j = head*F + f
__global__ void gat_edge_agg(const int* __restrict__ row, const int* __restrict__ col,
                             const float* __restrict__ ew, const float* __restrict__ s,
                             const float* __restrict__ h, float* __restrict__ out,
                             int total, int H, int F) {
  int i = blockIdx.x * blockDim.x + threadIdx.x;  // edge*(H*F) + j
  if (i >= total) return;
  const int HF = H * F;
  int e = i / HF, j = i % HF, head = j / F;
  int r = row[e], c = col[e];
  float att = ew[e * H + head] / (s[r * H + head] + GAT_EPS);
  atomicAdd(&out[(size_t)r * HF + j], att * h[(size_t)c * HF + j]);
}

__global__ void gat_bias_act(float* __restrict__ x, const float* __restrict__ b,
                             int total, int HF, int relu) {
  int i = blockIdx.x * blockDim.x + threadIdx.x;
  if (i >= total) return;
  float v = x[i] + b[i % HF];
  if (relu) v = v > 0.f ? v : 0.f;
  x[i] = v;
}

// ---------------- host ----------------

extern "C" void kernel_launch(void* const* d_in, const int* in_sizes, int n_in,
                              void* d_out, int out_size, void* d_ws, size_t ws_size,
                              hipStream_t stream) {
  const float* x   = (const float*)d_in[0];
  const int*   row = (const int*)d_in[1];
  const int*   col = (const int*)d_in[2];
  const float* W1  = (const float*)d_in[3];
  const float* b1  = (const float*)d_in[4];
  const float* al1 = (const float*)d_in[5];
  const float* ar1 = (const float*)d_in[6];
  const float* W2  = (const float*)d_in[7];
  const float* b2  = (const float*)d_in[8];
  const float* al2 = (const float*)d_in[9];
  const float* ar2 = (const float*)d_in[10];

  const int N = in_sizes[0] / 256;  // 50000
  const int E = in_sizes[1];        // 800000

  size_t off = 0;
  char* base = (char*)d_ws;
  auto bump = [&](size_t bytes) -> void* {
    void* p = base + off;
    off = (off + bytes + 255) & ~(size_t)255;
    return p;
  };

  __bf16* xbf  = (__bf16*)bump((size_t)N * 256 * 2);
  __bf16* W1t  = (__bf16*)bump((size_t)256 * 128 * 2);
  __bf16* W2t  = (__bf16*)bump((size_t)128 * 64 * 2);
  float*  h1   = (float*) bump((size_t)N * 128 * 4);
  float*  hl1  = (float*) bump((size_t)N * 4 * 4);
  float*  hr1  = (float*) bump((size_t)N * 4 * 4);
  int*    m1   = (int*)   bump((size_t)N * 4 * 4);
  float*  s1   = (float*) bump((size_t)N * 4 * 4);
  float*  ew1  = (float*) bump((size_t)E * 4 * 4);
  float*  agg1 = (float*) bump((size_t)N * 128 * 4);
  __bf16* h1bf = (__bf16*)bump((size_t)N * 128 * 2);
  float*  h2   = (float*) bump((size_t)N * 64 * 4);
  float*  hl2  = (float*) bump((size_t)N * 4);
  float*  hr2  = (float*) bump((size_t)N * 4);
  int*    m2   = (int*)   bump((size_t)N * 4);
  float*  s2   = (float*) bump((size_t)N * 4);
  float*  ew2  = (float*) bump((size_t)E * 4);
  (void)ws_size;

  const unsigned NEG_INF_ENC = 0x807FFFFFu;  // enc_f32(-inf)
  float* out = (float*)d_out;

  // ---- init ----
  gat_fill_u32<<<CDIV(N * 4, 256), 256, 0, stream>>>((unsigned*)m1, N * 4, NEG_INF_ENC);
  gat_fill_u32<<<CDIV(N * 4, 256), 256, 0, stream>>>((unsigned*)s1, N * 4, 0u);
  gat_fill_u32<<<CDIV(N * 128, 256), 256, 0, stream>>>((unsigned*)agg1, N * 128, 0u);
  gat_fill_u32<<<CDIV(N, 256), 256, 0, stream>>>((unsigned*)m2, N, NEG_INF_ENC);
  gat_fill_u32<<<CDIV(N, 256), 256, 0, stream>>>((unsigned*)s2, N, 0u);
  gat_fill_u32<<<CDIV(out_size, 256), 256, 0, stream>>>((unsigned*)d_out, out_size, 0u);

  // ---- precision demotion for WMMA ----
  gat_cvt_bf16<<<CDIV(N * 256, 256), 256, 0, stream>>>(x, xbf, N * 256);
  gat_cvt_transpose<<<CDIV(256 * 128, 256), 256, 0, stream>>>(W1, W1t, 256, 128);
  gat_cvt_transpose<<<CDIV(128 * 64, 256), 256, 0, stream>>>(W2, W2t, 128, 64);

  // ---- layer 1: h1 = x @ W1 (bf16 WMMA, f32 accum) ----
  {
    int strips = CDIV(N, 16);
    gat_wmma_gemm<8><<<CDIV(strips, 4), 128, 0, stream>>>(xbf, W1t, h1, N, 128, 256);
  }
  gat_proj_lr<<<CDIV(N * 4, 256), 256, 0, stream>>>(h1, al1, ar1, hl1, hr1, N * 4, 4, 32);
  gat_edge_max<<<CDIV(E * 4, 256), 256, 0, stream>>>(row, col, hl1, hr1, m1, E * 4, 4, 0.2f);
  gat_edge_exp<<<CDIV(E * 4, 256), 256, 0, stream>>>(row, col, hl1, hr1, m1, s1, ew1,
                                                     E * 4, 4, 0.2f);
  gat_edge_agg<<<CDIV(E * 128, 256), 256, 0, stream>>>(row, col, ew1, s1, h1, agg1,
                                                       E * 128, 4, 32);
  gat_bias_act<<<CDIV(N * 128, 256), 256, 0, stream>>>(agg1, b1, N * 128, 128, 1);

  // ---- layer 2: h2 = relu_out @ W2 ----
  gat_cvt_bf16<<<CDIV(N * 128, 256), 256, 0, stream>>>(agg1, h1bf, N * 128);
  {
    int strips = CDIV(N, 16);
    gat_wmma_gemm<4><<<CDIV(strips, 4), 128, 0, stream>>>(h1bf, W2t, h2, N, 64, 128);
  }
  gat_proj_lr<<<CDIV(N, 256), 256, 0, stream>>>(h2, al2, ar2, hl2, hr2, N, 1, 64);
  gat_edge_max<<<CDIV(E, 256), 256, 0, stream>>>(row, col, hl2, hr2, m2, E, 1, 0.2f);
  gat_edge_exp<<<CDIV(E, 256), 256, 0, stream>>>(row, col, hl2, hr2, m2, s2, ew2, E, 1, 0.2f);
  gat_edge_agg<<<CDIV(E * 64, 256), 256, 0, stream>>>(row, col, ew2, s2, h2, out,
                                                      E * 64, 1, 64);
  gat_bias_act<<<CDIV(N * 64, 256), 256, 0, stream>>>(out, b2, N * 64, 64, 0);
}
